// CausalSelfAttention_28939489640587
// MI455X (gfx1250) — matrix operand order repaired and checked by reading, not compiled, not run on hardware
//
#include <hip/hip_runtime.h>
#include <cstdint>

// ---------------------------------------------------------------------------
// CDNA5 (gfx1250) causal self-attention: bf16 WMMA GEMMs + flash attention.
// Data movement: TDM tensor_load_to_lds (TENSORcnt) for flash Q/K tiles,
// global_load_async_to_lds_b128 (ASYNCcnt) for GEMM A tiles.
// B=4, S=2048, D=1024, H=16, HD=64.
// ---------------------------------------------------------------------------

typedef __attribute__((ext_vector_type(16))) __bf16   bf16x16;
typedef __attribute__((ext_vector_type(8)))  __bf16   bf16x8;
typedef __attribute__((ext_vector_type(8)))  float    f32x8;
typedef __attribute__((ext_vector_type(4)))  uint32_t u32x4;
typedef __attribute__((ext_vector_type(8)))  uint32_t u32x8;

#define Bd 4
#define Sd 2048
#define Dd 1024
#define Hd 16
#define HDd 64

// Build a 16-element bf16 A/B fragment from two contiguous 16-byte LDS chunks.
__device__ __forceinline__ bf16x16 make_frag(const __bf16* p0, const __bf16* p1) {
    union { bf16x16 v; struct { bf16x8 lo; bf16x8 hi; } s; } u;
    u.s.lo = *(const bf16x8*)p0;
    u.s.hi = *(const bf16x8*)p1;
    return u.v;
}

// Async DMA of 16 bytes from global memory straight into LDS (per lane).
// Tracked by ASYNCcnt.
__device__ __forceinline__ void async_copy_b128(void* lds_dst, const void* gsrc) {
    uint32_t lds = (uint32_t)(uintptr_t)lds_dst;
    unsigned long long ga = (unsigned long long)(uintptr_t)gsrc;
    asm volatile("global_load_async_to_lds_b128 %0, %1, off"
                 :: "v"(lds), "v"(ga) : "memory");
}

__device__ __forceinline__ void wait_async_all() {
    asm volatile("s_wait_asynccnt 0x0" ::: "memory");
}

// ---------------------------------------------------------------------------
// Tensor Data Mover: DMA one 2D tile (64 x 64 bf16 elements) global -> LDS,
// row-major packed in LDS.  D# built per CDNA5 ISA 8.3/8.4; 2-group form
// (VADDR2/3 = NULL) is legal for tensors up to 2D.  Tracked by TENSORcnt.
// ---------------------------------------------------------------------------
__device__ __forceinline__ void tdm_load_tile64x64_bf16(uint32_t lds_off,
                                                        const void* gtile,
                                                        uint32_t tensor_rows,
                                                        uint32_t row_stride_elems) {
    uint64_t ga = (uint64_t)(uintptr_t)gtile;
    u32x4 g0;
    g0[0] = 0x1u;                                      // count=1, user descriptor
    g0[1] = lds_off;                                   // lds_addr (bytes)
    g0[2] = (uint32_t)ga;                              // global_addr[31:0]
    g0[3] = (uint32_t)((ga >> 32) & 0x1FFFFFFu)        // global_addr[56:32]
            | (2u << 30);                              // type = 2 ("image")
    u32x8 g1;
    g1[0] = (1u << 16);                                // data_size = 1 (2 bytes)
    g1[1] = (row_stride_elems & 0xFFFFu) << 16;        // tensor_dim0[15:0]
    g1[2] = (row_stride_elems >> 16)                   // tensor_dim0[31:16]
            | ((tensor_rows & 0xFFFFu) << 16);         // tensor_dim1[15:0]
    g1[3] = (tensor_rows >> 16)                        // tensor_dim1[31:16]
            | (64u << 16);                             // tile_dim0 = 64
    g1[4] = 64u;                                       // tile_dim1 = 64, tile_dim2 = 0
    g1[5] = row_stride_elems;                          // tensor_dim0_stride[31:0]
    g1[6] = 0u;                                        // stride0[47:32] | stride1[15:0]
    g1[7] = 0u;
    asm volatile("tensor_load_to_lds %0, %1" :: "s"(g0), "s"(g1) : "memory");
}

__device__ __forceinline__ void wait_tensor_all() {
    __builtin_amdgcn_s_wait_tensorcnt(0);
}

// ---------------------------------------------------------------------------
// fp32 -> bf16 conversion (vectorized x4)
// ---------------------------------------------------------------------------
__global__ void convert_f32_bf16_kernel(const float* __restrict__ in,
                                        __bf16* __restrict__ out, int n4) {
    int i = blockIdx.x * blockDim.x + threadIdx.x;
    if (i < n4) {
        float4 f = *(const float4*)(in + (size_t)i * 4);
        union { uint2 u; __bf16 b[4]; } t;
        t.b[0] = (__bf16)f.x; t.b[1] = (__bf16)f.y;
        t.b[2] = (__bf16)f.z; t.b[3] = (__bf16)f.w;
        *(uint2*)(out + (size_t)i * 4) = t.u;
    }
}

// ---------------------------------------------------------------------------
// Tiled bf16 GEMM with f32 accumulate:  C[M,N] = A[M,K] * B[K,N] + bias[N]
// Block = 256 threads (8 waves), tile 128x128, BK=32.
// A tile staged with async global->LDS DMA; B staged transposed (Bs[n][k]).
// ---------------------------------------------------------------------------
template <bool OUT_BF16>
__global__ __launch_bounds__(256)
void gemm_bf16_kernel(const __bf16* __restrict__ A, const __bf16* __restrict__ Bw,
                      const float* __restrict__ bias,
                      __bf16* __restrict__ outb, float* __restrict__ outf,
                      int M, int N, int K) {
    __shared__ __attribute__((aligned(16))) __bf16 As[128 * 32];
    __shared__ __attribute__((aligned(16))) __bf16 Bs[128 * 32];

    const int tid  = threadIdx.x;
    const int lane = tid & 31;
    const int w    = tid >> 5;          // wave 0..7
    const int wm   = w >> 2;            // 0..1  (rows)
    const int wn   = w & 3;             // 0..3  (cols)
    const int ln   = lane & 15;
    const int hf   = lane >> 4;
    const int m0   = blockIdx.y * 128;
    const int n0   = blockIdx.x * 128;

    f32x8 acc[4][2] = {};

    for (int k0 = 0; k0 < K; k0 += 32) {
        __syncthreads();
        // stage A tile 128x32 row-major: 512 x 16B async DMA chunks
        for (int i = tid; i < 512; i += 256) {
            int r  = i >> 2;
            int c8 = (i & 3) * 8;
            async_copy_b128(As + r * 32 + c8,
                            A + (size_t)(m0 + r) * K + k0 + c8);
        }
        // stage B tile 32x128 transposed into Bs[n][k]
        for (int i = tid; i < 128 * 32 / 4; i += 256) {
            int kk = i >> 5;
            int n4 = (i & 31) * 4;
            union { uint2 u; __bf16 b[4]; } t;
            t.u = *(const uint2*)(Bw + (size_t)(k0 + kk) * N + n0 + n4);
#pragma unroll
            for (int j = 0; j < 4; ++j) Bs[(n4 + j) * 32 + kk] = t.b[j];
        }
        // prefetch next K-slab into cache (global_prefetch_b8)
        if (k0 + 32 < K) {
            __builtin_prefetch(A + (size_t)(m0 + (tid >> 1)) * K + k0 + 32 + (tid & 1) * 16, 0, 1);
            __builtin_prefetch(Bw + (size_t)(k0 + 32 + (tid >> 3)) * N + n0 + (tid & 7) * 16, 0, 1);
        }
        wait_async_all();
        __syncthreads();

        bf16x16 bfrag[2];
#pragma unroll
        for (int tj = 0; tj < 2; ++tj) {
            const __bf16* bp = Bs + (wn * 32 + tj * 16 + ln) * 32;
            bfrag[tj] = make_frag(bp + hf * 8, bp + 16 + hf * 8);
        }
#pragma unroll
        for (int ti = 0; ti < 4; ++ti) {
            const __bf16* ap = As + (wm * 64 + ti * 16 + ln) * 32;
            bf16x16 afrag = make_frag(ap + hf * 8, ap + 16 + hf * 8);
#pragma unroll
            for (int tj = 0; tj < 2; ++tj) {
                acc[ti][tj] = __builtin_amdgcn_wmma_f32_16x16x32_bf16(
                    false, afrag, false, bfrag[tj], (short)0, acc[ti][tj], false, false);
            }
        }
    }

    // epilogue: C/D layout row = v + 8*hf, col = lane&15
#pragma unroll
    for (int tj = 0; tj < 2; ++tj) {
        int c = n0 + wn * 32 + tj * 16 + ln;
        float bv = bias[c];
#pragma unroll
        for (int ti = 0; ti < 4; ++ti) {
            int rbase = m0 + wm * 64 + ti * 16 + hf * 8;
#pragma unroll
            for (int v = 0; v < 8; ++v) {
                float val = acc[ti][tj][v] + bv;
                size_t off = (size_t)(rbase + v) * N + c;
                if constexpr (OUT_BF16) outb[off] = (__bf16)val;
                else                    outf[off] = val;
            }
        }
    }
}

// ---------------------------------------------------------------------------
// Flash attention over qkv (bf16, layout [B*S, 3D], Q|K|V at col 0/D/2D).
// Grid: (S/64, B*H).  128 threads = 4 waves; each wave owns 16 q-rows.
// Q/K tiles DMAed by the Tensor Data Mover (wave 0 issues, TENSORcnt);
// V staged transposed manually (TDM cannot transpose), overlapping the TDM.
// ---------------------------------------------------------------------------
__global__ __launch_bounds__(128)
void flash_attn_kernel(const __bf16* __restrict__ qkv, __bf16* __restrict__ y) {
    __shared__ __attribute__((aligned(16))) __bf16 Qs[64 * 64];
    __shared__ __attribute__((aligned(16))) __bf16 Ks[64 * 64];
    __shared__ __attribute__((aligned(16))) __bf16 Vt[64 * 64];
    __shared__ __attribute__((aligned(16))) __bf16 Pbuf[4 * 16 * 64];

    const int tid  = threadIdx.x;
    const int lane = tid & 31;
    const int w    = tid >> 5;
    const int ln   = lane & 15;
    const int hf   = lane >> 4;
    const int bh   = blockIdx.y;            // 0..63
    const int bb   = bh >> 4;
    const int h    = bh & 15;
    const int qt   = blockIdx.x;            // 0..31
    const size_t rowbase = (size_t)bb * Sd;
    const int q_base = qt * 64;
    const int m0     = w * 16;              // wave's q-row stripe
    const float scale = 0.125f;             // 1/sqrt(64)
    const float NEG = -1e30f;

    // stage Q tile (64x64 bf16) via TDM
    if (w == 0) {
        tdm_load_tile64x64_bf16((uint32_t)(uintptr_t)Qs,
                                qkv + (rowbase + q_base) * (3 * Dd) + h * HDd,
                                (uint32_t)(Bd * Sd), 3 * Dd);
        wait_tensor_all();
    }
    __syncthreads();

    // Q fragments are loop-invariant: build once (kc = 0, 32)
    bf16x16 afragQ[2];
#pragma unroll
    for (int kq = 0; kq < 2; ++kq) {
        const __bf16* qp = Qs + (m0 + ln) * 64 + kq * 32;
        afragQ[kq] = make_frag(qp + hf * 8, qp + 16 + hf * 8);
    }

    f32x8 acc_o[4] = {};
    float mst[8], lst[8];
#pragma unroll
    for (int v = 0; v < 8; ++v) { mst[v] = NEG; lst[v] = 0.f; }

    __bf16* Pw = Pbuf + w * 16 * 64;

    for (int kt = 0; kt <= qt; ++kt) {
        const int k_base = kt * 64;
        __syncthreads();
        // K tile: one TDM descriptor, issued by wave 0, runs async
        if (w == 0) {
            tdm_load_tile64x64_bf16((uint32_t)(uintptr_t)Ks,
                                    qkv + (rowbase + k_base) * (3 * Dd) + Dd + h * HDd,
                                    (uint32_t)(Bd * Sd), 3 * Dd);
        }
        // V tile transposed into Vt[hd][key] (overlaps the TDM transfer)
        for (int i = tid; i < 64 * 16; i += 128) {
            int r  = i >> 4;
            int d4 = (i & 15) * 4;
            union { uint2 u; __bf16 b[4]; } t;
            t.u = *(const uint2*)(qkv + (rowbase + k_base + r) * (3 * Dd) + 2 * Dd + h * HDd + d4);
#pragma unroll
            for (int j = 0; j < 4; ++j) Vt[(d4 + j) * 64 + r] = t.b[j];
        }
        if (w == 0) wait_tensor_all();
        __syncthreads();

        // scores S = Q K^T   (16 q-rows x 64 keys per wave)
        f32x8 sc[4] = {};
#pragma unroll
        for (int kq = 0; kq < 2; ++kq) {
            int kc = kq * 32;
#pragma unroll
            for (int nt = 0; nt < 4; ++nt) {
                const __bf16* kp = Ks + (nt * 16 + ln) * 64 + kc;
                bf16x16 bfrag = make_frag(kp + hf * 8, kp + 16 + hf * 8);
                sc[nt] = __builtin_amdgcn_wmma_f32_16x16x32_bf16(
                    false, afragQ[kq], false, bfrag, (short)0, sc[nt], false, false);
            }
        }

        const bool diag = (kt == qt);
#pragma unroll
        for (int v = 0; v < 8; ++v) {
            int rloc = m0 + v + hf * 8;          // local q row of this element
            float s[4];
            float mv = NEG;
#pragma unroll
            for (int nt = 0; nt < 4; ++nt) {
                float sv = sc[nt][v] * scale;
                sv = (diag && (nt * 16 + ln) > rloc) ? NEG : sv;
                s[nt] = sv;
                mv = fmaxf(mv, sv);
            }
            // row-max across the 16 lanes holding this row (stays in half)
            mv = fmaxf(mv, __shfl_xor(mv, 1, 32));
            mv = fmaxf(mv, __shfl_xor(mv, 2, 32));
            mv = fmaxf(mv, __shfl_xor(mv, 4, 32));
            mv = fmaxf(mv, __shfl_xor(mv, 8, 32));

            float mnew  = fmaxf(mst[v], mv);
            float alpha = __expf(mst[v] - mnew);
            float psum  = 0.f;
#pragma unroll
            for (int nt = 0; nt < 4; ++nt) {
                float p = __expf(s[nt] - mnew);
                psum += p;
                Pw[(v + hf * 8) * 64 + nt * 16 + ln] = (__bf16)p;
            }
            psum += __shfl_xor(psum, 1, 32);
            psum += __shfl_xor(psum, 2, 32);
            psum += __shfl_xor(psum, 4, 32);
            psum += __shfl_xor(psum, 8, 32);

            lst[v] = lst[v] * alpha + psum;
            mst[v] = mnew;
#pragma unroll
            for (int ot = 0; ot < 4; ++ot) acc_o[ot][v] *= alpha;
        }
        __syncthreads();   // P visible; also orders P stores vs loads

        // O += P V
#pragma unroll
        for (int kc = 0; kc < 64; kc += 32) {
            const __bf16* pp = Pw + ln * 64 + kc;
            bf16x16 afrag = make_frag(pp + hf * 8, pp + 16 + hf * 8);
#pragma unroll
            for (int ot = 0; ot < 4; ++ot) {
                const __bf16* vp = Vt + (ot * 16 + ln) * 64 + kc;
                bf16x16 bfrag = make_frag(vp + hf * 8, vp + 16 + hf * 8);
                acc_o[ot] = __builtin_amdgcn_wmma_f32_16x16x32_bf16(
                    false, afrag, false, bfrag, (short)0, acc_o[ot], false, false);
            }
        }
    }

    // normalize and write y (bf16, [B*S, D] with head offset)
#pragma unroll
    for (int ot = 0; ot < 4; ++ot) {
        int c = h * HDd + ot * 16 + ln;
#pragma unroll
        for (int v = 0; v < 8; ++v) {
            float o = acc_o[ot][v] / lst[v];
            y[(rowbase + q_base + m0 + v + hf * 8) * Dd + c] = (__bf16)o;
        }
    }
}

// ---------------------------------------------------------------------------
// Launch
// ---------------------------------------------------------------------------
extern "C" void kernel_launch(void* const* d_in, const int* in_sizes, int n_in,
                              void* d_out, int out_size, void* d_ws, size_t ws_size,
                              hipStream_t stream) {
    const float* x      = (const float*)d_in[0];
    const float* W_attn = (const float*)d_in[1];
    const float* b_attn = (const float*)d_in[2];
    const float* W_proj = (const float*)d_in[3];
    const float* b_proj = (const float*)d_in[4];
    float* out = (float*)d_out;

    const size_t M = (size_t)Bd * Sd;        // 8192 rows

    char* p = (char*)d_ws;
    __bf16* xb   = (__bf16*)p;  p += M * Dd * 2;                 // 16 MB
    __bf16* Wab  = (__bf16*)p;  p += (size_t)Dd * 3 * Dd * 2;    //  6 MB
    __bf16* Wpb  = (__bf16*)p;  p += (size_t)Dd * Dd * 2;        //  2 MB
    __bf16* qkvb = (__bf16*)p;  p += M * 3 * Dd * 2;             // 48 MB
    __bf16* yb   = (__bf16*)p;                                   // 16 MB

    // fp32 -> bf16 conversions
    {
        int n4 = (int)(M * Dd / 4);
        convert_f32_bf16_kernel<<<(n4 + 255) / 256, 256, 0, stream>>>(x, xb, n4);
        n4 = Dd * 3 * Dd / 4;
        convert_f32_bf16_kernel<<<(n4 + 255) / 256, 256, 0, stream>>>(W_attn, Wab, n4);
        n4 = Dd * Dd / 4;
        convert_f32_bf16_kernel<<<(n4 + 255) / 256, 256, 0, stream>>>(W_proj, Wpb, n4);
    }

    // qkv = x @ W_attn + b_attn  (bf16 out)
    gemm_bf16_kernel<true><<<dim3(3 * Dd / 128, (int)(M / 128)), 256, 0, stream>>>(
        xb, Wab, b_attn, qkvb, nullptr, (int)M, 3 * Dd, Dd);

    // flash attention -> y (bf16)
    flash_attn_kernel<<<dim3(Sd / 64, Bd * Hd), 128, 0, stream>>>(qkvb, yb);

    // out = y @ W_proj + b_proj  (fp32 out)
    gemm_bf16_kernel<false><<<dim3(Dd / 128, (int)(M / 128)), 256, 0, stream>>>(
        yb, Wpb, b_proj, nullptr, out, (int)M, Dd, Dd);
}
